// SwinBlock_29953101922765
// MI455X (gfx1250) — matrix operand order repaired and checked by reading, not compiled, not run on hardware
//
#include <hip/hip_runtime.h>
#include <hip/hip_bf16.h>
#include <cstddef>

#define DIM      192
#define NHEADS   6
#define HD       32
#define WSZ      7
#define SSH      3
#define NTOK     49
#define NPAD     64
#define HP       56
#define MLPH     768
#define QKSCALE  0.17677669529663687f   /* 32^-0.5 */

typedef __attribute__((ext_vector_type(16))) _Float16 v16h;
typedef __attribute__((ext_vector_type(8)))  _Float16 v8h;
typedef __attribute__((ext_vector_type(8)))  float    v8f;

// ---------------------------------------------------------------------------
// WMMA fragment loaders (wave32, V_WMMA_F32_16X16X32_F16 layouts)
// A 16x32 f16: lane m = lane&15 holds row m; lane group (lane>>4) selects the
// K+16 half; v16h elems 0..15 = K[g*16 .. g*16+15] (contiguous).
// Two contiguous 16B loads per lane.
__device__ inline v16h frag_a(const _Float16* __restrict__ base, int stride) {
  const int lane = threadIdx.x & 31;
  const _Float16* p = base + (lane & 15) * stride + (lane >> 4) * 16;
  v8h lo = *(const v8h*)(p);
  v8h hi = *(const v8h*)(p + 8);
  v16h r;
#pragma unroll
  for (int i = 0; i < 8; ++i) { r[i] = lo[i]; r[i + 8] = hi[i]; }
  return r;
}

// B 32x16 f16 from column-major Bt[n][k] (stride = K length):
// lane n = lane&15 holds column n; lane group selects K+16; the 16 halves are
// K-contiguous -> one 32B load per lane.
__device__ inline v16h frag_b(const _Float16* __restrict__ bt, int stride) {
  const int lane = threadIdx.x & 31;
  const _Float16* p = bt + (lane & 15) * stride + (lane >> 4) * 16;
  return *(const v16h*)p;
}

__device__ inline v8f wmma_f16(v16h a, v16h b, v8f c) {
  return __builtin_amdgcn_wmma_f32_16x16x32_f16(false, a, false, b, (short)0, c,
                                                false, false);
}

// ---------------------------------------------------------------------------
// Weight conversion: src [R][C] f32 row-major -> dst [C][R] f16 (out-major,
// in-contiguous) so frag_b reads are contiguous.
__global__ __launch_bounds__(256) void transpose_to_f16(
    const float* __restrict__ src, _Float16* __restrict__ dst, int R, int C) {
  int idx = blockIdx.x * 256 + threadIdx.x;
  if (idx < R * C) {
    int r = idx / C, c = idx % C;
    dst[c * R + r] = (_Float16)src[r * C + c];
  }
}

__device__ inline int region_id(int ry, int rx) {
  const int a = (ry < HP - WSZ) ? 0 : (ry < HP - SSH) ? 1 : 2;
  const int b = (rx < HP - WSZ) ? 0 : (rx < HP - SSH) ? 1 : 2;
  return a * 3 + b;
}

// ---------------------------------------------------------------------------
// Fused LN1 + shifted-window attention + proj + residual.  One workgroup
// (256 threads = 8 wave32) per window; 2048 windows.
// Dynamic LDS layout (122880 B):
//   sA  64x192 f16 (LN1'd window input; reused as sO after QKV)
//   sQ  64x192 f16 (pre-scaled), sK 64x192 f16, sVt 192x64 f16
//   sS  64x64 f32 (logits), sP 64x64 f16 (softmax probs)
__global__ __launch_bounds__(256) void attn_kernel(
    const float* __restrict__ x, const float* __restrict__ qkv_b,
    const float* __restrict__ proj_b, const float* __restrict__ rpb,
    const float* __restrict__ ln1g, const float* __restrict__ ln1b,
    const _Float16* __restrict__ qkv_wt, const _Float16* __restrict__ proj_wt,
    float* __restrict__ hbuf) {
  extern __shared__ char smem[];
  _Float16* sA  = (_Float16*)smem;          // 24576 B (also sO)
  _Float16* sQ  = sA + NPAD * DIM;          // 24576 B
  _Float16* sK  = sQ + NPAD * DIM;          // 24576 B
  _Float16* sVt = sK + NPAD * DIM;          // 192x64, 24576 B
  float*    sS  = (float*)(sVt + DIM * NPAD);   // 16384 B
  _Float16* sP  = (_Float16*)(sS + NPAD * NPAD); // 8192 B

  const int w    = blockIdx.x;
  const int b    = w >> 6;
  const int wi   = w & 63;
  const int wy   = wi >> 3, wx = wi & 7;
  const int tid  = threadIdx.x;
  const int wave = tid >> 5;
  const int lane = tid & 31;

  // ---- LN1 + gather shifted window rows into sA (f16) ----
  for (int r = wave; r < NPAD; r += 8) {
    if (r < NTOK) {
      const int iy = r / WSZ, ix = r % WSZ;
      const int ys = (wy * WSZ + iy + SSH) % HP;
      const int xs = (wx * WSZ + ix + SSH) % HP;
      const float* row = x + ((size_t)b * 3136 + (size_t)(ys * 56 + xs)) * DIM;
      float v[6];
      float s = 0.f;
#pragma unroll
      for (int i = 0; i < 6; ++i) { v[i] = row[lane + 32 * i]; s += v[i]; }
#pragma unroll
      for (int o = 16; o > 0; o >>= 1) s += __shfl_xor(s, o, 32);
      const float mu = s * (1.f / DIM);
      float q = 0.f;
#pragma unroll
      for (int i = 0; i < 6; ++i) { float d = v[i] - mu; q += d * d; }
#pragma unroll
      for (int o = 16; o > 0; o >>= 1) q += __shfl_xor(q, o, 32);
      const float rs = rsqrtf(q * (1.f / DIM) + 1e-5f);
#pragma unroll
      for (int i = 0; i < 6; ++i) {
        const int c = lane + 32 * i;
        sA[r * DIM + c] = (_Float16)((v[i] - mu) * rs * ln1g[c] + ln1b[c]);
      }
    } else {
      for (int c = lane; c < DIM; c += 32) sA[r * DIM + c] = (_Float16)0.f;
    }
  }
  __syncthreads();

  // ---- QKV GEMM: [64x192] @ [192x576], tiles 4(M) x 36(N), K = 6 steps ----
  for (int nt = wave; nt < 36; nt += 8) {
    // Scalarize the tile column base: uniform per wave -> SGPR branches below.
    const int nbase = __builtin_amdgcn_readfirstlane(nt * 16);
    // Prefetch next weight tile this wave will consume (global_prefetch_b8).
    if (nt + 8 < 36)
      __builtin_prefetch(qkv_wt + (size_t)(nbase + 128) * DIM +
                             (size_t)lane * DIM, 0, 1);
    for (int mt = 0; mt < 4; ++mt) {
      v8f acc;
      const float bias = qkv_b[nbase + (lane & 15)];
#pragma unroll
      for (int i = 0; i < 8; ++i) acc[i] = bias;
#pragma unroll
      for (int kk = 0; kk < 6; ++kk) {
        v16h a  = frag_a(sA + mt * 16 * DIM + kk * 32, DIM);
        v16h bb = frag_b(qkv_wt + (size_t)nbase * DIM + kk * 32, DIM);
        acc = wmma_f16(a, bb, acc);
      }
      const int n = lane & 15, mo = (lane >> 4) * 8;
      // Tile never straddles the Q/K/V boundaries (192 % 16 == 0), so route
      // the whole tile with one scalar branch on nbase.
      if (nbase < DIM) {
        const int col = nbase + n;
#pragma unroll
        for (int i = 0; i < 8; ++i)
          sQ[(mt * 16 + mo + i) * DIM + col] = (_Float16)(acc[i] * QKSCALE);
      } else if (nbase < 2 * DIM) {
        const int col = nbase - DIM + n;
#pragma unroll
        for (int i = 0; i < 8; ++i)
          sK[(mt * 16 + mo + i) * DIM + col] = (_Float16)acc[i];
      } else {
        const int vr = nbase - 2 * DIM + n;
#pragma unroll
        for (int i = 0; i < 8; ++i)
          sVt[vr * NPAD + (mt * 16 + mo + i)] = (_Float16)acc[i];
      }
    }
  }
  __syncthreads();

  // ---- per-head attention ----
  for (int h = 0; h < NHEADS; ++h) {
    // S = Q Kt : 4x4 tiles, K = 32 (single WMMA); +rpb +shift mask
    for (int t = wave; t < 16; t += 8) {
      const int mt = t >> 2, nt = t & 3;
      v8f acc;
#pragma unroll
      for (int i = 0; i < 8; ++i) acc[i] = 0.f;
      v16h a  = frag_a(sQ + mt * 16 * DIM + h * HD, DIM);
      v16h bb = frag_b(sK + nt * 16 * DIM + h * HD, DIM);
      acc = wmma_f16(a, bb, acc);
      const int n = lane & 15, mo = (lane >> 4) * 8;
#pragma unroll
      for (int i = 0; i < 8; ++i) {
        const int gi = mt * 16 + mo + i;   // query token
        const int gj = nt * 16 + n;        // key token
        float vv = acc[i];
        if (gi < NTOK && gj < NTOK) {
          const int yi = gi / WSZ, xi = gi % WSZ;
          const int yj = gj / WSZ, xj = gj % WSZ;
          vv += rpb[((yi - yj + WSZ - 1) * (2 * WSZ - 1) + (xi - xj + WSZ - 1)) * NHEADS + h];
          const int ri = region_id(wy * WSZ + yi, wx * WSZ + xi);
          const int rj = region_id(wy * WSZ + yj, wx * WSZ + xj);
          if (ri != rj) vv -= 100.f;
        } else {
          vv = -1e9f;
        }
        sS[gi * NPAD + gj] = vv;
      }
    }
    __syncthreads();

    // softmax: one thread per row
    if (tid < NPAD) {
      float mx = -1e9f;
      for (int j = 0; j < NTOK; ++j) mx = fmaxf(mx, sS[tid * NPAD + j]);
      float sum = 0.f;
      for (int j = 0; j < NTOK; ++j) sum += __expf(sS[tid * NPAD + j] - mx);
      const float inv = 1.f / sum;
      for (int j = 0; j < NTOK; ++j)
        sP[tid * NPAD + j] = (_Float16)(__expf(sS[tid * NPAD + j] - mx) * inv);
      for (int j = NTOK; j < NPAD; ++j) sP[tid * NPAD + j] = (_Float16)0.f;
    }
    __syncthreads();

    // O_h = P @ V_h : 4x2 tiles, one per wave; K = 2 steps. sA reused as sO.
    {
      const int mt = wave >> 1, nt = wave & 1;
      v8f acc;
#pragma unroll
      for (int i = 0; i < 8; ++i) acc[i] = 0.f;
#pragma unroll
      for (int kk = 0; kk < 2; ++kk) {
        v16h a  = frag_a(sP + mt * 16 * NPAD + kk * 32, NPAD);
        v16h bb = frag_b(sVt + (h * HD + nt * 16) * NPAD + kk * 32, NPAD);
        acc = wmma_f16(a, bb, acc);
      }
      const int n = lane & 15, mo = (lane >> 4) * 8;
#pragma unroll
      for (int i = 0; i < 8; ++i)
        sA[(mt * 16 + mo + i) * DIM + h * HD + nt * 16 + n] = (_Float16)acc[i];
    }
    __syncthreads();
  }

  // ---- proj GEMM [64x192]@[192x192] + shortcut, inverse-roll scatter ----
  for (int t = wave; t < 48; t += 8) {
    const int mt = t / 12, nt = t % 12;
    const int nbase = __builtin_amdgcn_readfirstlane(nt * 16);
    v8f acc;
    const float bias = proj_b[nbase + (lane & 15)];
#pragma unroll
    for (int i = 0; i < 8; ++i) acc[i] = bias;
#pragma unroll
    for (int kk = 0; kk < 6; ++kk) {
      v16h a  = frag_a(sA + mt * 16 * DIM + kk * 32, DIM);
      v16h bb = frag_b(proj_wt + (size_t)nbase * DIM + kk * 32, DIM);
      acc = wmma_f16(a, bb, acc);
    }
    const int n = lane & 15, mo = (lane >> 4) * 8;
#pragma unroll
    for (int i = 0; i < 8; ++i) {
      const int gi = mt * 16 + mo + i;
      if (gi < NTOK) {
        const int iy = gi / WSZ, ix = gi % WSZ;
        const int yo = (wy * WSZ + iy + SSH) % HP;
        const int xo = (wx * WSZ + ix + SSH) % HP;
        const size_t idx =
            ((size_t)b * 3136 + (size_t)(yo * 56 + xo)) * DIM + nbase + n;
        hbuf[idx] = x[idx] + acc[i];
      }
    }
  }
}

// ---------------------------------------------------------------------------
// Fused LN2 + MLP (fc1 + exact GELU + fc2) + residual.  One workgroup per 64
// rows (100352 = 1568*64).  LDS: sA 64x192 f16 + sH 64x768 f16 = 122880 B.
__global__ __launch_bounds__(256) void mlp_kernel(
    const float* __restrict__ hbuf, const float* __restrict__ ln2g,
    const float* __restrict__ ln2b, const float* __restrict__ b1,
    const float* __restrict__ b2, const _Float16* __restrict__ w1t,
    const _Float16* __restrict__ w2t, float* __restrict__ out) {
  extern __shared__ char smem[];
  _Float16* sA = (_Float16*)smem;      // 64x192
  _Float16* sH = sA + 64 * DIM;        // 64x768

  const size_t rowbase = (size_t)blockIdx.x * 64;
  const int tid = threadIdx.x;
  const int wave = tid >> 5;
  const int lane = tid & 31;

  // LN2 rows -> sA
  for (int r = wave; r < 64; r += 8) {
    const float* row = hbuf + (rowbase + r) * DIM;
    float v[6];
    float s = 0.f;
#pragma unroll
    for (int i = 0; i < 6; ++i) { v[i] = row[lane + 32 * i]; s += v[i]; }
#pragma unroll
    for (int o = 16; o > 0; o >>= 1) s += __shfl_xor(s, o, 32);
    const float mu = s * (1.f / DIM);
    float q = 0.f;
#pragma unroll
    for (int i = 0; i < 6; ++i) { float d = v[i] - mu; q += d * d; }
#pragma unroll
    for (int o = 16; o > 0; o >>= 1) q += __shfl_xor(q, o, 32);
    const float rs = rsqrtf(q * (1.f / DIM) + 1e-5f);
#pragma unroll
    for (int i = 0; i < 6; ++i) {
      const int c = lane + 32 * i;
      sA[r * DIM + c] = (_Float16)((v[i] - mu) * rs * ln2g[c] + ln2b[c]);
    }
  }
  __syncthreads();

  // fc1 + GELU: [64x192]@[192x768], tiles 4x48, K = 6 steps
  for (int nt = wave; nt < 48; nt += 8) {
    const int nbase = __builtin_amdgcn_readfirstlane(nt * 16);
    if (nt + 8 < 48)
      __builtin_prefetch(w1t + (size_t)(nbase + 128) * DIM +
                             (size_t)lane * DIM, 0, 1);
    for (int mt = 0; mt < 4; ++mt) {
      v8f acc;
      const float bias = b1[nbase + (lane & 15)];
#pragma unroll
      for (int i = 0; i < 8; ++i) acc[i] = bias;
#pragma unroll
      for (int kk = 0; kk < 6; ++kk) {
        v16h a  = frag_a(sA + mt * 16 * DIM + kk * 32, DIM);
        v16h bb = frag_b(w1t + (size_t)nbase * DIM + kk * 32, DIM);
        acc = wmma_f16(a, bb, acc);
      }
      const int n = lane & 15, mo = (lane >> 4) * 8;
#pragma unroll
      for (int i = 0; i < 8; ++i) {
        float vv = acc[i];
        vv = 0.5f * vv * (1.f + erff(vv * 0.70710678118654752f));  // exact GELU
        sH[(mt * 16 + mo + i) * MLPH + nbase + n] = (_Float16)vv;
      }
    }
  }
  __syncthreads();

  // fc2 + residual: [64x768]@[768x192], tiles 4x12, K = 24 steps
  for (int t = wave; t < 48; t += 8) {
    const int mt = t / 12, nt = t % 12;
    const int nbase = __builtin_amdgcn_readfirstlane(nt * 16);
    v8f acc;
    const float bias = b2[nbase + (lane & 15)];
#pragma unroll
    for (int i = 0; i < 8; ++i) acc[i] = bias;
#pragma unroll
    for (int kk = 0; kk < 24; ++kk) {
      v16h a  = frag_a(sH + mt * 16 * MLPH + kk * 32, MLPH);
      v16h bb = frag_b(w2t + (size_t)nbase * MLPH + kk * 32, MLPH);
      acc = wmma_f16(a, bb, acc);
    }
    const int n = lane & 15, mo = (lane >> 4) * 8;
#pragma unroll
    for (int i = 0; i < 8; ++i) {
      const size_t idx = (rowbase + mt * 16 + mo + i) * DIM + nbase + n;
      out[idx] = hbuf[idx] + acc[i];
    }
  }
}

// ---------------------------------------------------------------------------
extern "C" void kernel_launch(void* const* d_in, const int* in_sizes, int n_in,
                              void* d_out, int out_size, void* d_ws,
                              size_t ws_size, hipStream_t stream) {
  const float* x      = (const float*)d_in[0];
  const float* qkv_w  = (const float*)d_in[1];
  const float* qkv_b  = (const float*)d_in[2];
  const float* proj_w = (const float*)d_in[3];
  const float* proj_b = (const float*)d_in[4];
  const float* rpb    = (const float*)d_in[5];
  const float* ln1g   = (const float*)d_in[6];
  const float* ln1b   = (const float*)d_in[7];
  const float* ln2g   = (const float*)d_in[8];
  const float* ln2b   = (const float*)d_in[9];
  const float* w1     = (const float*)d_in[10];
  const float* b1     = (const float*)d_in[11];
  const float* w2     = (const float*)d_in[12];
  const float* b2     = (const float*)d_in[13];
  float* out = (float*)d_out;

  char* ws = (char*)d_ws;
  _Float16* qkv_wt  = (_Float16*)ws; ws += (size_t)576 * 192 * 2;
  _Float16* proj_wt = (_Float16*)ws; ws += (size_t)192 * 192 * 2;
  _Float16* w1t     = (_Float16*)ws; ws += (size_t)768 * 192 * 2;
  _Float16* w2t     = (_Float16*)ws; ws += (size_t)192 * 768 * 2;
  float* hbuf = (float*)ws;  // 32*3136*192 f32 = 77,070,336 B

  const int kSmem = 122880;
  (void)hipFuncSetAttribute((const void*)attn_kernel,
                            hipFuncAttributeMaxDynamicSharedMemorySize, kSmem);
  (void)hipFuncSetAttribute((const void*)mlp_kernel,
                            hipFuncAttributeMaxDynamicSharedMemorySize, kSmem);

  transpose_to_f16<<<(192 * 576 + 255) / 256, 256, 0, stream>>>(qkv_w, qkv_wt, 192, 576);
  transpose_to_f16<<<(192 * 192 + 255) / 256, 256, 0, stream>>>(proj_w, proj_wt, 192, 192);
  transpose_to_f16<<<(192 * 768 + 255) / 256, 256, 0, stream>>>(w1, w1t, 192, 768);
  transpose_to_f16<<<(768 * 192 + 255) / 256, 256, 0, stream>>>(w2, w2t, 768, 192);

  attn_kernel<<<2048, 256, kSmem, stream>>>(x, qkv_b, proj_b, rpb, ln1g, ln1b,
                                            qkv_wt, proj_wt, hbuf);
  mlp_kernel<<<1568, 256, kSmem, stream>>>(hbuf, ln2g, ln2b, b1, b2, w1t, w2t,
                                           out);
}
